// VectorQuantizer_61143154426545
// MI455X (gfx1250) — compile-verified, matrix operand order checked by reading, not run on hardware
//
#include <hip/hip_runtime.h>

#define NUM_EMBED 1024
#define EMBED_DIM 256
#define N_VEC     32768                 // 32*32*32 rows
#define NE        8388608               // N_VEC * EMBED_DIM

typedef __attribute__((ext_vector_type(16))) _Float16 v16h;
typedef __attribute__((ext_vector_type(8)))  _Float16 v8h;
typedef __attribute__((ext_vector_type(8)))  float    v8f;

// ---------------------------------------------------------------- init
__global__ void vq_init(float* acc) { acc[0] = 0.f; acc[1] = 0.f; }

// ----------------------------------------------- copy x->out, x->f16 (transposed), sum(x^2)
__global__ __launch_bounds__(256)
void vq_prep_x(const float* __restrict__ x, float* __restrict__ out,
               _Float16* __restrict__ xh, float* __restrict__ accum) {
  __shared__ float tile[32][33];
  __shared__ float red[256];
  int bid   = blockIdx.x;
  int b     = bid >> 8;        // 0..31
  int rem   = bid & 255;
  int ctile = rem >> 5;        // 0..7   (channel tile, 32 channels each)
  int wtile = rem & 31;        // 0..31  (hw tile, 32 positions each)
  int tid   = threadIdx.x;
  int lrow  = tid >> 5;        // 0..7
  int lcol  = tid & 31;        // 0..31

  float ss = 0.f;
  size_t base = (size_t)b * (EMBED_DIM * 1024) + (size_t)(ctile * 32) * 1024 + (size_t)(wtile * 32);
  #pragma unroll
  for (int i = 0; i < 4; ++i) {
    int c_local = lrow + i * 8;
    size_t g = base + (size_t)c_local * 1024 + lcol;
    float v = x[g];
    out[g] = v;                        // out == x (faithful to reference "bug")
    tile[c_local][lcol] = v;
    ss += v * v;
  }
  __syncthreads();
  #pragma unroll
  for (int i = 0; i < 4; ++i) {
    int hw_local = lrow + i * 8;
    int n = b * 1024 + wtile * 32 + hw_local;
    xh[(size_t)n * EMBED_DIM + ctile * 32 + lcol] = (_Float16)tile[lcol][hw_local];
  }
  red[tid] = ss;
  __syncthreads();
  for (int s = 128; s > 0; s >>= 1) {
    if (tid < s) red[tid] += red[tid + s];
    __syncthreads();
  }
  if (tid == 0) atomicAdd(&accum[0], red[0]);
}

// ----------------------------------------------- codebook -> f16 + ||e||^2 (f32)
__global__ __launch_bounds__(256)
void vq_prep_cb(const float* __restrict__ cb, _Float16* __restrict__ cbh,
                float* __restrict__ cbnorm) {
  __shared__ float red[256];
  int k = blockIdx.x;
  int t = threadIdx.x;
  float v = cb[(size_t)k * EMBED_DIM + t];
  cbh[(size_t)k * EMBED_DIM + t] = (_Float16)v;
  red[t] = v * v;
  __syncthreads();
  for (int s = 128; s > 0; s >>= 1) {
    if (t < s) red[t] += red[t + s];
    __syncthreads();
  }
  if (t == 0) cbnorm[k] = red[0];
}

// ----------------------------------------------- WMMA score GEMM + row-min reduce
#define LB_STRIDE 264   // 256 halves + 16B pad -> dword stride 132, conflict-free ds_load_b128

__global__ __launch_bounds__(256)
void vq_gemm_min(const _Float16* __restrict__ xh, const _Float16* __restrict__ cbh,
                 const float* __restrict__ cbnorm, float* __restrict__ accum) {
  __shared__ _Float16 lB[2][16 * LB_STRIDE];   // double-buffered B tile (2 x 8.25 KB)
  int tid  = threadIdx.x;
  int wave = tid >> 5;
  int lane = tid & 31;
  int m    = lane & 15;     // row (A) / entry (B) within 16-wide tile
  int hi   = lane >> 4;     // K-half selector per ISA 16-bit layout
  int rowBase = blockIdx.x * 128 + wave * 16;

  // staging coordinates: 256 threads move 16 entries x 256 halves as 512 v8h chunks
  int se   = tid >> 5;        // entry 0..7 (and +8)
  int soff = (tid & 31) * 8;  // half-offset within entry row

  // A-tile (16 rows x 256 K, f16) held in registers for the whole column sweep
  const _Float16* arow = xh + (size_t)(rowBase + m) * EMBED_DIM;
  v16h a[8];
  #pragma unroll
  for (int j = 0; j < 8; ++j) {
    int kb = j * 32;
    v8h lo  = *(const v8h*)(arow + kb + hi * 8);
    v8h hi8 = *(const v8h*)(arow + kb + 16 + hi * 8);
    #pragma unroll
    for (int t = 0; t < 8; ++t) { a[j][t] = lo[t]; a[j][8 + t] = hi8[t]; }
  }

  float runMin[8];
  #pragma unroll
  for (int i = 0; i < 8; ++i) runMin[i] = 3.4e38f;

  // stage column tile 0 into buffer 0
  *(v8h*)(&lB[0][se * LB_STRIDE + soff]) =
      *(const v8h*)(cbh + (size_t)se * EMBED_DIM + soff);
  *(v8h*)(&lB[0][(se + 8) * LB_STRIDE + soff]) =
      *(const v8h*)(cbh + (size_t)(se + 8) * EMBED_DIM + soff);
  __syncthreads();

  for (int ct = 0; ct < 64; ++ct) {
    int buf = ct & 1;

    // issue global loads for the NEXT tile first (LOADcnt: doesn't block the DS->WMMA chain)
    v8h n0, n1;
    if (ct + 1 < 64) {
      const _Float16* s2 = cbh + (size_t)(ct + 1) * 16 * EMBED_DIM;
      n0 = *(const v8h*)(s2 + (size_t)se * EMBED_DIM + soff);
      n1 = *(const v8h*)(s2 + (size_t)(se + 8) * EMBED_DIM + soff);
    }

    // preload the full B tile (64 VGPRs): 16 ds_load_b128, single dscnt wait
    const _Float16* brow = &lB[buf][(size_t)m * LB_STRIDE];
    v16h bfrag[8];
    #pragma unroll
    for (int j = 0; j < 8; ++j) {
      int kb = j * 32;
      v8h blo = *(const v8h*)(brow + kb + hi * 8);
      v8h bhi = *(const v8h*)(brow + kb + 16 + hi * 8);
      #pragma unroll
      for (int t = 0; t < 8; ++t) { bfrag[j][t] = blo[t]; bfrag[j][8 + t] = bhi[t]; }
    }

    // 8 back-to-back WMMAs accumulating the 16x16 dot tile
    v8f acc = {};
    #pragma unroll
    for (int j = 0; j < 8; ++j)
      acc = __builtin_amdgcn_wmma_f32_16x16x32_f16(false, a[j], false, bfrag[j],
                                                   (short)0, acc, false, false);

    // score = ||e||^2 - 2*dot ; each lane owns entry n = ct*16 + m, 8 rows in acc
    float cn = cbnorm[ct * 16 + m];
    #pragma unroll
    for (int i = 0; i < 8; ++i) {
      float s = __builtin_fmaf(-2.f, acc[i], cn);
      runMin[i] = fminf(runMin[i], s);
    }

    // commit the staged tile into the other buffer; single barrier per iteration
    if (ct + 1 < 64) {
      *(v8h*)(&lB[buf ^ 1][se * LB_STRIDE + soff]) = n0;
      *(v8h*)(&lB[buf ^ 1][(se + 8) * LB_STRIDE + soff]) = n1;
    }
    __syncthreads();
  }

  // C-layout: VGPR i -> row (i + 8*hi), 16 lanes of a half share a row.
  float rowSum = 0.f;
  #pragma unroll
  for (int i = 0; i < 8; ++i) {
    float v = runMin[i];
    #pragma unroll
    for (int mask = 8; mask >= 1; mask >>= 1)
      v = fminf(v, __shfl_xor(v, mask, 32));
    rowSum += v;   // lanes 0 / 16 now hold the sum of mins for rows 0..7 / 8..15
  }
  if (m == 0) atomicAdd(&accum[1], rowSum);
}

// ----------------------------------------------- scalar loss
__global__ void vq_finalize(const float* __restrict__ accum, float* __restrict__ out) {
  out[NE] = 1.25f * (accum[0] + accum[1]) / (float)NE;
}

// ================================================================ launch
extern "C" void kernel_launch(void* const* d_in, const int* in_sizes, int n_in,
                              void* d_out, int out_size, void* d_ws, size_t ws_size,
                              hipStream_t stream) {
  const float* x  = (const float*)d_in[0];   // [32,256,32,32]
  const float* cb = (const float*)d_in[1];   // [1024,256]
  float* out = (float*)d_out;                // NE floats + 1 loss scalar

  char* ws = (char*)d_ws;
  float*    accum  = (float*)ws;                                   // 2 f32 accumulators
  _Float16* xh     = (_Float16*)(ws + 256);                        // 16 MB  f16 [N][E]
  _Float16* cbh    = (_Float16*)(ws + 256 + (size_t)NE * 2);       // 512 KB f16 [K][E]
  float*    cbnorm = (float*)(ws + 256 + (size_t)NE * 2
                              + (size_t)NUM_EMBED * EMBED_DIM * 2);// 4 KB

  vq_init<<<1, 1, 0, stream>>>(accum);
  vq_prep_x<<<8192, 256, 0, stream>>>(x, out, xh, accum);
  vq_prep_cb<<<NUM_EMBED, 256, 0, stream>>>(cb, cbh, cbnorm);
  vq_gemm_min<<<N_VEC / 128, 256, 0, stream>>>(xh, cbh, cbnorm, accum);
  vq_finalize<<<1, 1, 0, stream>>>(accum, out);
}
